// DTNetv0_42296837931603
// MI455X (gfx1250) — matrix operand order repaired and checked by reading, not compiled, not run on hardware
//
#include <hip/hip_runtime.h>
#include <hip/hip_bf16.h>
#include <stdint.h>

#define B_TOTAL   131072
#define IN_DIM    128
#define N_NODES   511
#define N_NODES_P 512
#define N_LEAVES  512
#define N_ACTIONS 16
#define K2        1024      // padded 2*N_NODES
#define TILE_M    64

typedef __attribute__((ext_vector_type(16))) __bf16      v16bf;
typedef __attribute__((ext_vector_type(8)))  float       v8f;
typedef __attribute__((ext_vector_type(4)))  unsigned int u32x4;

union Frag { u32x4 u[2]; v16bf v; };

__device__ __forceinline__ unsigned short f2bf(float f) {
    unsigned int u = __float_as_uint(f);
    u += 0x7FFFu + ((u >> 16) & 1u);          // round-to-nearest-even (prep only)
    return (unsigned short)(u >> 16);
}

// two f32 -> packed bf16x2 in 3 VALU ops: round-to-nearest via +0x8000 on the
// raw bits, then one v_perm_b32 extracts both high halves.
__device__ __forceinline__ unsigned int pack2bf(float lo, float hi) {
    unsigned int ul = __float_as_uint(lo) + 0x8000u;
    unsigned int uh = __float_as_uint(hi) + 0x8000u;
    return __builtin_amdgcn_perm(uh, ul, 0x07060302u);  // {uh[31:16], ul[31:16]}
}

// order-preserving float -> u32 key (monotone for all non-NaN floats)
__device__ __forceinline__ unsigned int fkey(float f) {
    unsigned int u = __float_as_uint(f);
    return u ^ (unsigned int)(((int)u >> 31) | 0x80000000);
}
__device__ __forceinline__ float funkey(unsigned int k) {
    unsigned int u = (k & 0x80000000u) ? (k ^ 0x80000000u) : ~k;
    return __uint_as_float(u);
}

// ---------------- weight conversion: fp32 -> padded bf16 -----------------
__global__ __launch_bounds__(512)
void dtnet_prep(const float* __restrict__ W1, const float* __restrict__ b1,
                const float* __restrict__ W2,
                unsigned short* __restrict__ W1bf,   // [512][128]
                unsigned short* __restrict__ W2bf,   // [512][1024]
                float* __restrict__ b1p)             // [512]
{
    const int T1 = N_NODES_P * IN_DIM;       // 65536
    const int T2 = T1 + N_LEAVES * K2;       // +524288
    const int T3 = T2 + N_NODES_P;           // +512
    for (int i = blockIdx.x * blockDim.x + threadIdx.x; i < T3;
         i += gridDim.x * blockDim.x) {
        if (i < T1) {
            int n = i >> 7, k = i & 127;
            W1bf[i] = (n < N_NODES) ? f2bf(W1[n * IN_DIM + k]) : (unsigned short)0;
        } else if (i < T2) {
            int j = i - T1;
            int l = j >> 10, k = j & 1023;
            float v = 0.0f;
            if (k < N_NODES)                 v = W2[l * (2 * N_NODES) + k];
            else if (k >= 512 && k < 512 + N_NODES)
                                             v = W2[l * (2 * N_NODES) + N_NODES + (k - 512)];
            W2bf[j] = f2bf(v);
        } else {
            int n = i - T2;
            b1p[n] = (n < N_NODES) ? b1[n] : 0.0f;
        }
    }
}

// ---------------- fused forward: 64 batch rows per block -----------------
__global__ __launch_bounds__(256)
void dtnet_main(const float* __restrict__ x,
                const float* __restrict__ b1p,
                const unsigned short* __restrict__ W1bf,
                const unsigned short* __restrict__ W2bf,
                const int* __restrict__ leaf_actions,
                float* __restrict__ out)
{
    __shared__ __align__(16) unsigned char smem[16384 + 131072 + 2048 + 4096];
    unsigned short* xbf  = (unsigned short*)smem;                    // [64][128] bf16
    unsigned short* zS   = (unsigned short*)(smem + 16384);          // [64][1024] bf16
    float*          yS   = (float*)(smem + 16384);                   // overlay [64][512] f32
    int*            actS = (int*)(smem + 16384 + 131072);            // [512]
    unsigned int*   poolK= (unsigned int*)(smem + 16384 + 131072 + 2048); // [64][16]

    const int tid    = threadIdx.x;
    const int lane   = tid & 31;
    const int wave   = tid >> 5;
    const int laneLo = lane & 15;
    const bool hiH   = lane >= 16;
    const int rowBase = blockIdx.x * TILE_M;

    // ---- stage x tile as bf16, stage leaf_actions, init pooled keys ----
    {
        const float2* xsrc = (const float2*)(x + (size_t)rowBase * IN_DIM);
        unsigned int* dst = (unsigned int*)xbf;
        for (int p = tid; p < TILE_M * IN_DIM / 2; p += 256) {
            float2 v = xsrc[p];
            dst[p] = pack2bf(v.x, v.y);
        }
        for (int i = tid; i < N_LEAVES; i += 256) actS[i] = leaf_actions[i];
        for (int i = tid; i < TILE_M * N_ACTIONS; i += 256) poolK[i] = 0u;
    }
    __syncthreads();

    // ---- GEMM1: h = x @ W1^T + b1 ; z = [relu(h), relu(-h)] -> LDS ----
    {
        const int mtile = wave & 3;      // 4 row-tiles of 16
        const int nhalf = wave >> 2;     // 2 halves of 256 nodes
        Frag A[4];
        {
            const unsigned char* rowp = (const unsigned char*)xbf
                + (size_t)(mtile * 16 + laneLo) * (IN_DIM * 2);
            #pragma unroll
            for (int kb = 0; kb < 4; ++kb) {
                const unsigned char* p = rowp + kb * 64 + (hiH ? 16 : 0);
                A[kb].u[0] = *(const u32x4*)(p);
                A[kb].u[1] = *(const u32x4*)(p + 32);
            }
        }
        #pragma unroll 1
        for (int nt = 0; nt < 16; ++nt) {
            const int col = nhalf * 256 + nt * 16 + laneLo;   // node id (lane-resident N)
            v8f acc = {};
            #pragma unroll
            for (int kb = 0; kb < 4; ++kb) {
                Frag Bf;
                const unsigned char* p = (const unsigned char*)W1bf
                    + (size_t)col * (IN_DIM * 2) + kb * 64 + (hiH ? 16 : 0);
                Bf.u[0] = *(const u32x4*)(p);
                Bf.u[1] = *(const u32x4*)(p + 32);
                acc = __builtin_amdgcn_wmma_f32_16x16x32_bf16(
                          false, A[kb].v, false, Bf.v, (short)0, acc, false, false);
            }
            const float bv = b1p[col];
            // adjacent-M accumulator pairs -> one v_perm-packed bf16x2; halves
            // store to consecutive LDS rows (ds_store_b16 / _d16_hi)
            #pragma unroll
            for (int r = 0; r < 4; ++r) {
                const int row0 = mtile * 16 + 2 * r + (hiH ? 8 : 0);
                const float h0 = acc[2 * r] + bv;
                const float h1 = acc[2 * r + 1] + bv;
                const unsigned int zp = pack2bf(fmaxf(h0, 0.0f), fmaxf(h1, 0.0f));
                const unsigned int zn = pack2bf(fmaxf(-h0, 0.0f), fmaxf(-h1, 0.0f));
                zS[row0 * K2 + col]             = (unsigned short)zp;
                zS[(row0 + 1) * K2 + col]       = (unsigned short)(zp >> 16);
                zS[row0 * K2 + col + 512]       = (unsigned short)zn;
                zS[(row0 + 1) * K2 + col + 512] = (unsigned short)(zn >> 16);
            }
        }
    }
    __syncthreads();

    // ---- GEMM2: y = z @ W2^T  (K = 1024, each wave owns 64 leaves x 64 rows) ----
    v8f acc[4][4];
    #pragma unroll
    for (int mt = 0; mt < 4; ++mt)
        #pragma unroll
        for (int nt = 0; nt < 4; ++nt) acc[mt][nt] = (v8f){};

    const int leafBase = wave * 64;
    #pragma unroll 1
    for (int kb = 0; kb < 32; ++kb) {
        Frag A[4];
        #pragma unroll
        for (int mt = 0; mt < 4; ++mt) {
            const unsigned char* p = (const unsigned char*)zS
                + (size_t)(mt * 16 + laneLo) * (K2 * 2) + kb * 64 + (hiH ? 16 : 0);
            A[mt].u[0] = *(const u32x4*)(p);
            A[mt].u[1] = *(const u32x4*)(p + 32);
        }
        #pragma unroll
        for (int nt = 0; nt < 4; ++nt) {
            Frag Bf;
            const unsigned char* p = (const unsigned char*)W2bf
                + (size_t)(leafBase + nt * 16 + laneLo) * (K2 * 2)
                + kb * 64 + (hiH ? 16 : 0);
            Bf.u[0] = *(const u32x4*)(p);
            Bf.u[1] = *(const u32x4*)(p + 32);
            #pragma unroll
            for (int mt = 0; mt < 4; ++mt)
                acc[mt][nt] = __builtin_amdgcn_wmma_f32_16x16x32_bf16(
                                  false, A[mt].v, false, Bf.v, (short)0,
                                  acc[mt][nt], false, false);
        }
    }
    __syncthreads();          // everyone done reading zS before overlaying with y

    #pragma unroll
    for (int mt = 0; mt < 4; ++mt)
        #pragma unroll
        for (int nt = 0; nt < 4; ++nt)
            #pragma unroll
            for (int r = 0; r < 8; ++r) {
                const int row  = mt * 16 + r + (hiH ? 8 : 0);
                const int colL = leafBase + nt * 16 + laneLo;
                yS[row * N_LEAVES + colL] = acc[mt][nt][r];
            }
    __syncthreads();

    // ---- grouped segment-max via LDS atomic max on order-preserving keys ----
    {
        #pragma unroll
        for (int j = 0; j < 2; ++j) {
            const int l = tid * 2 + j;                 // 2 leaves per thread
            const int a = actS[l];
            #pragma unroll 4
            for (int row = 0; row < TILE_M; ++row) {
                const unsigned int key = fkey(yS[row * N_LEAVES + l]);
                atomicMax(&poolK[row * N_ACTIONS + a], key);   // ds_max_u32
            }
        }
    }
    __syncthreads();

    // ---- softmax over 16 actions, one thread per row ----
    if (tid < TILE_M) {
        float v[16], mx = -__builtin_inff();
        #pragma unroll
        for (int a = 0; a < 16; ++a) {
            v[a] = funkey(poolK[tid * N_ACTIONS + a]);
            mx = fmaxf(mx, v[a]);
        }
        float s = 0.0f;
        #pragma unroll
        for (int a = 0; a < 16; ++a) { v[a] = __expf(v[a] - mx); s += v[a]; }
        const float inv = 1.0f / s;
        float* o = out + (size_t)(rowBase + tid) * N_ACTIONS;
        #pragma unroll
        for (int a = 0; a < 16; ++a) o[a] = v[a] * inv;
    }
}

extern "C" void kernel_launch(void* const* d_in, const int* in_sizes, int n_in,
                              void* d_out, int out_size, void* d_ws, size_t ws_size,
                              hipStream_t stream)
{
    const float* x  = (const float*)d_in[0];
    const float* W1 = (const float*)d_in[1];
    const float* b1 = (const float*)d_in[2];
    const float* W2 = (const float*)d_in[3];
    const int*   la = (const int*)d_in[4];

    unsigned short* W1bf = (unsigned short*)d_ws;                           // 131072 B
    unsigned short* W2bf = (unsigned short*)((char*)d_ws + 131072);         // 1048576 B
    float*          b1p  = (float*)((char*)d_ws + 131072 + 1048576);        // 2048 B

    dtnet_prep<<<1153, 512, 0, stream>>>(W1, b1, W2, W1bf, W2bf, b1p);
    dtnet_main<<<B_TOTAL / TILE_M, 256, 0, stream>>>(x, b1p, W1bf, W2bf, la,
                                                     (float*)d_out);
}
